// StyleGAN2Generator_58909771432212
// MI455X (gfx1250) — matrix operand order repaired and checked by reading, not compiled, not run on hardware
//
#include <hip/hip_runtime.h>
#include <stdint.h>
#include <stddef.h>

// ---------------------------------------------------------------------------
// StyleGAN2 generator forward for MI455X (gfx1250, wave32, WMMA).
// - mapping MLP, modulated 3x3 convs, ToRGB 1x1 convs all run on
//   v_wmma_f32_16x16x32_bf16
// - style s folded into the activations (pre-scaled in place per layer),
//   demod folded into the epilogue -> each conv is a dense implicit GEMM
// - conv input tiles staged to LDS via the Tensor Data Mover
//   (tensor_load_to_lds + s_wait_tensorcnt); arity dispatched on clang major
//   (5-arg on ROCm7.2/clang-22, 6-arg on amdgpu-toolchain/clang-23+)
// ---------------------------------------------------------------------------

typedef __attribute__((ext_vector_type(16))) __bf16 v16bf;
typedef __attribute__((ext_vector_type(8)))  __bf16 v8bf;
typedef __attribute__((ext_vector_type(8)))  float  v8f;
typedef unsigned int u32x4 __attribute__((ext_vector_type(4)));
typedef int          i32x4 __attribute__((ext_vector_type(4)));
typedef int          i32x8 __attribute__((ext_vector_type(8)));

#define NLAYERS 13
#define WDIM    512
#define BATCH   16
#define CICHUNK 32
#define TILEMAX 768   // max R*W elements staged per input channel

#if __has_builtin(__builtin_amdgcn_tensor_load_to_lds)
#define HAVE_TDM 1
#else
#define HAVE_TDM 0
#endif

#if __has_builtin(__builtin_amdgcn_s_wait_tensorcnt)
#define WAIT_TENSOR() __builtin_amdgcn_s_wait_tensorcnt(0)
#else
#define WAIT_TENSOR()
#endif

// ----------------------------- helpers -------------------------------------

__device__ __forceinline__ float hash_normal(uint32_t x) {
  // deterministic Gaussian stand-in for randn noise (no runtime check exists)
  x ^= x >> 17; x *= 0xed5ad4bbu;
  x ^= x >> 11; x *= 0xac4c1b51u;
  x ^= x >> 15; x *= 0x31848babu;
  x ^= x >> 14;
  float u1 = ((float)(x & 0xFFFFu) + 1.0f) * (1.0f / 65537.0f);
  float u2 = (float)(x >> 16) * (1.0f / 65536.0f);
  return sqrtf(-2.0f * __logf(u1)) * __cosf(6.28318530718f * u2);
}

// ------------------------- mapping network ---------------------------------
// One workgroup (8 waves); w[16x512] ping-pongs through LDS in bf16.

__global__ void __launch_bounds__(256) mapping_kernel(
    const float* __restrict__ z, const float* __restrict__ map_w,
    const float* __restrict__ map_b, float* __restrict__ wout)
{
  __shared__ __bf16 bufA[16 * WDIM];
  __shared__ __bf16 bufB[16 * WDIM];
  __shared__ float  red[256];
  __shared__ float  rinv[16];

  const int t    = threadIdx.x;
  const int lane = t & 31;
  const int wave = t >> 5;
  const int lane15 = lane & 15;
  const bool lo  = (lane < 16);

  { // normalize z rows into bufA (bf16)
    const int r = t >> 4, c = t & 15;
    float sum = 0.f;
    for (int j = 0; j < 32; ++j) {
      float v = z[r * WDIM + c + 16 * j];
      sum += v * v;
    }
    red[t] = sum;
    __syncthreads();
    if (t < 16) {
      float s2 = 0.f;
      for (int j = 0; j < 16; ++j) s2 += red[t * 16 + j];
      rinv[t] = 1.0f / fmaxf(sqrtf(s2), 1e-12f);
    }
    __syncthreads();
    for (int idx = t; idx < 16 * WDIM; idx += 256) {
      int rr = idx >> 9;
      bufA[idx] = (__bf16)(z[idx] * rinv[rr]);
    }
    __syncthreads();
  }

  __bf16* cur = bufA;
  __bf16* nxt = bufB;
  const int khalfA = lo ? 0 : 8;
  const int kbB    = lo ? 0 : 8;

  for (int l = 0; l < 8; ++l) {
    const float* Wl = map_w + (size_t)l * WDIM * WDIM;
    const float* bl = map_b + (size_t)l * WDIM;
    for (int ntl = 0; ntl < 4; ++ntl) {
      const int nt = wave * 4 + ntl;
      const int n  = nt * 16 + lane15;
      v8f acc = {0.f, 0.f, 0.f, 0.f, 0.f, 0.f, 0.f, 0.f};
      for (int ks = 0; ks < WDIM / 32; ++ks) {
        const int k0 = ks * 32;
        v16bf a;
#pragma unroll
        for (int e = 0; e < 16; ++e) {
          const int klocal = (e < 8) ? (khalfA + e) : (8 + khalfA + e);
          a[e] = cur[lane15 * WDIM + k0 + klocal];
        }
        v16bf bv;
#pragma unroll
        for (int e = 0; e < 16; ++e)
          bv[e] = (__bf16)Wl[(size_t)n * WDIM + k0 + kbB + e + ((e < 8) ? 0 : 8)];
        acc = __builtin_amdgcn_wmma_f32_16x16x32_bf16(
            false, a, false, bv, (short)0, acc, false, false);
      }
      const float bsv = bl[n];
#pragma unroll
      for (int r = 0; r < 8; ++r) {
        const int m = r + (lo ? 0 : 8);
        float v = acc[r] + bsv;
        v = (v > 0.f) ? v : 0.2f * v;
        nxt[m * WDIM + n] = (__bf16)v;
        if (l == 7) wout[m * WDIM + n] = v;
      }
    }
    __syncthreads();
    __bf16* tmp = cur; cur = nxt; nxt = tmp;
  }
}

// --------------------------- small kernels ---------------------------------

__global__ void style_kernel(const float* __restrict__ w,
                             const float* __restrict__ mw,
                             const float* __restrict__ mb,
                             float* __restrict__ s, int Ci)
{
  int i = blockIdx.x * blockDim.x + threadIdx.x;
  if (i >= BATCH * Ci) return;
  int b = i / Ci, c = i - b * Ci;
  const float* wr = w + (size_t)b * WDIM;
  const float* mr = mw + (size_t)c * WDIM;
  float acc = mb[c];
  for (int k = 0; k < WDIM; ++k) acc += wr[k] * mr[k];
  s[i] = acc;
}

__global__ void prep_weights(const float* __restrict__ Wf,
                             __bf16* __restrict__ Wb,
                             float* __restrict__ WS,
                             int Co, int Ci, int kk2)
{
  int i = blockIdx.x * blockDim.x + threadIdx.x;
  if (i >= Co * Ci) return;
  const float* src = Wf + (size_t)i * kk2;
  __bf16* dst = Wb + (size_t)i * kk2;
  float sq = 0.f;
  for (int j = 0; j < kk2; ++j) {
    float v = src[j];
    sq += v * v;
    dst[j] = (__bf16)v;
  }
  if (WS) WS[i] = sq;
}

__global__ void demod_kernel(const float* __restrict__ WS,
                             const float* __restrict__ s,
                             float* __restrict__ d, int Co, int Ci)
{
  int i = blockIdx.x * blockDim.x + threadIdx.x;
  if (i >= BATCH * Co) return;
  int b = i / Co, co = i - b * Co;
  const float* sr = s + (size_t)b * Ci;
  const float* wr = WS + (size_t)co * Ci;
  float acc = 0.f;
  for (int c = 0; c < Ci; ++c) { float sv = sr[c]; acc += wr[c] * sv * sv; }
  d[i] = rsqrtf(acc + 1e-8f);
}

__global__ void bcast_const(const float* __restrict__ cst,
                            __bf16* __restrict__ x, int CHW)
{
  int i = blockIdx.x * blockDim.x + threadIdx.x;
  if (i >= BATCH * CHW) return;
  x[i] = (__bf16)cst[i % CHW];
}

// in-place x[b,ci,:,:] *= s[b,ci]; buffer is dead after the conv consumes it
__global__ void scale_input(__bf16* __restrict__ x, const float* __restrict__ s,
                            int Ci, int HW)
{
  size_t i = (size_t)blockIdx.x * blockDim.x + threadIdx.x;
  size_t total = (size_t)BATCH * Ci * HW;
  if (i >= total) return;
  float sv = s[i / HW];   // index == b*Ci + ci
  x[i] = (__bf16)((float)x[i] * sv);
}

template <typename T>
__global__ void upsample2x(const T* __restrict__ in, T* __restrict__ out,
                           int C, int H, int W)
{
  const int OH = 2 * H, OW = 2 * W;
  size_t i = (size_t)blockIdx.x * blockDim.x + threadIdx.x;
  size_t total = (size_t)BATCH * C * OH * OW;
  if (i >= total) return;
  int ox = (int)(i % OW); size_t tq = i / OW;
  int oy = (int)(tq % OH); tq /= OH;
  int c = (int)(tq % C);  int b = (int)(tq / C);
  float sy = 0.5f * oy - 0.25f, sx = 0.5f * ox - 0.25f;
  int y0 = (int)floorf(sy), x0 = (int)floorf(sx);
  float fy = sy - y0, fx = sx - x0;
  int y0c = max(y0, 0), y1c = min(y0 + 1, H - 1);
  int x0c = max(x0, 0), x1c = min(x0 + 1, W - 1);
  const T* p = in + (((size_t)b * C + c) * H) * W;
  float v00 = (float)p[(size_t)y0c * W + x0c];
  float v01 = (float)p[(size_t)y0c * W + x1c];
  float v10 = (float)p[(size_t)y1c * W + x0c];
  float v11 = (float)p[(size_t)y1c * W + x1c];
  float v = (1.f - fy) * ((1.f - fx) * v00 + fx * v01) +
            fy * ((1.f - fx) * v10 + fx * v11);
  out[i] = (T)v;
}

// ------------------- modulated 3x3 conv (LDS + TDM + WMMA) -----------------
// Workgroup = 8 waves = 128 consecutive output positions x one 16-co tile.
// K loop runs in 32-channel chunks (288 k = 9 aligned 16x16x32 WMMA steps);
// each chunk's input rows (y-halo included) are staged into LDS by the TDM.

__global__ void __launch_bounds__(256) modconv_wmma(
    const __bf16* __restrict__ xs,   // pre-scaled input [B,Cin,H,W]
    const __bf16* __restrict__ wgt,  // packed [Co, Cin*9] bf16
    const float* __restrict__ dmod, const float* __restrict__ bias,
    const float* __restrict__ noise_w, int layer,
    __bf16* __restrict__ xout,
    int Cin, int Co, int H, int W, int mblocks)
{
  __shared__ __bf16 smem[CICHUNK * TILEMAX];   // 48 KB

  const int t = threadIdx.x;
  const int lane = t & 31;
  const int wave = t >> 5;
  const int lane15 = lane & 15;
  const bool lo = (lane < 16);

  const int HW = H * W;
  const int mtiles = HW >> 4;
  const int ntiles = (Co + 15) >> 4;

  int blk = blockIdx.x;
  const int b = blk / (ntiles * mblocks);
  blk -= b * ntiles * mblocks;
  const int nt = blk / mblocks;
  const int mb = blk - nt * mblocks;

  const int mt = mb * 8 + wave;
  const bool mvalid = (mt < mtiles);

  // staged row range (with 3x3 halo, clamped)
  const int p0 = mb * 128;
  const int pLast = min(p0 + 127, HW - 1);
  const int yLo = max(p0 / W - 1, 0);
  const int yHi = min(pLast / W + 1, H - 1);
  const int R = yHi - yLo + 1;
  const int RW = R * W;

  // B operand: column = spatial position
  const int p = mvalid ? (mt * 16 + lane15) : 0;
  const int y0 = p / W;
  const int x0 = p - y0 * W;
  const int kbB = lo ? 0 : 8;

  // A operand: row = output channel (contiguous bf16 weight chunks)
  const int coA = nt * 16 + lane15;
  const bool coAok = (coA < Co);
  const int K = Cin * 9;
  const __bf16* wrow = wgt + (size_t)(coAok ? coA : 0) * K;
  const int khalfA = lo ? 0 : 8;

  const __bf16* xb = xs + (size_t)b * Cin * HW;

  v8f acc = {0.f, 0.f, 0.f, 0.f, 0.f, 0.f, 0.f, 0.f};

  for (int c0 = 0; c0 < Cin; c0 += CICHUNK) {
#if HAVE_TDM
    if (t == 0) {
      // ---- Tensor DMA descriptor (CDNA5 ISA) ----
      uint64_t ga = (uint64_t)(uintptr_t)(xb + (size_t)c0 * HW +
                                          (size_t)yLo * W);
      uint32_t ldsoff = (uint32_t)(uintptr_t)(void*)&smem[0];
      u32x4 g0;
      g0[0] = 1u;                                   // count=1, user mode
      g0[1] = ldsoff;                               // lds_addr
      g0[2] = (uint32_t)ga;                         // global_addr[31:0]
      g0[3] = (uint32_t)((ga >> 32) & 0x1FFFFFFu) | (2u << 30); // addr hi|type=2
      uint32_t d0 = (uint32_t)W;                    // tensor_dim0
      uint32_t d1 = (uint32_t)(H - yLo);            // tensor_dim1 (rows left)
      uint64_t s0 = (uint64_t)W;                    // dim0 stride (row pitch)
      uint64_t s1 = (uint64_t)HW;                   // dim1 stride (chan pitch)
      i32x8 g1;
      g1[0] = (int)(1u << 16);                      // data_size=2B, no flags
      g1[1] = (int)((d0 & 0xFFFFu) << 16);          // dim0[15:0]
      g1[2] = (int)(((d0 >> 16) & 0xFFFFu) | ((d1 & 0xFFFFu) << 16));
      g1[3] = (int)(((d1 >> 16) & 0xFFFFu) | ((uint32_t)W << 16)); // tile_dim0
      g1[4] = (int)(((uint32_t)R & 0xFFFFu) | ((uint32_t)CICHUNK << 16));
      g1[5] = (int)(uint32_t)(s0 & 0xFFFFFFFFu);
      g1[6] = (int)((uint32_t)((s0 >> 32) & 0xFFFFu) |
                    ((uint32_t)(s1 & 0xFFFFu) << 16));
      g1[7] = (int)(uint32_t)((s1 >> 16) & 0xFFFFFFFFu);
      i32x4 g2;
      g2[0] = (int)(uint32_t)(Cin - c0);            // tensor_dim2 (z OOB -> 0)
      g2[1] = 0; g2[2] = 0; g2[3] = 0;
      i32x4 g3 = {0, 0, 0, 0};
#if defined(__clang_major__) && (__clang_major__ >= 23)
      i32x8 g4 = {0, 0, 0, 0, 0, 0, 0, 0};
      __builtin_amdgcn_tensor_load_to_lds(g0, g1, g2, g3, g4, 0);
#else
      __builtin_amdgcn_tensor_load_to_lds(g0, g1, g2, g3, 0);
#endif
    }
    WAIT_TENSOR();
#else
    for (int idx = t; idx < CICHUNK * RW; idx += 256) {
      int cl = idx / RW;
      int rem = idx - cl * RW;
      float v = 0.f;
      if (c0 + cl < Cin) v = (float)xb[(size_t)(c0 + cl) * HW + yLo * W + rem];
      smem[idx] = (__bf16)v;
    }
#endif
    __syncthreads();

    if (mvalid) {
      if (coAok) __builtin_prefetch(wrow + (c0 + CICHUNK) * 9, 0, 1);
      const int kLim = min(K - c0 * 9, CICHUNK * 9);  // k's in this chunk
      const int ksteps = (kLim + 31) >> 5;
#pragma unroll 3
      for (int ks = 0; ks < ksteps; ++ks) {
        const int kc0 = ks << 5;        // k offset within chunk
        const int kg0 = c0 * 9 + kc0;   // global k
        v16bf a;
        if (coAok && (kg0 + 32 <= K)) {
          // fast path: two aligned 16B runs per lane (K*2 is 16B-multiple)
          v8bf a0 = *(const v8bf*)(wrow + kg0 + khalfA);
          v8bf a1 = *(const v8bf*)(wrow + kg0 + 16 + khalfA);
#pragma unroll
          for (int e = 0; e < 8; ++e) { a[e] = a0[e]; a[8 + e] = a1[e]; }
        } else {
#pragma unroll
          for (int e = 0; e < 16; ++e) {
            const int klocal = (e < 8) ? (khalfA + e) : (8 + khalfA + e);
            const int kg = kg0 + klocal;
            float v = 0.f;
            if (coAok && kg < K) v = (float)wrow[kg];
            a[e] = (__bf16)v;
          }
        }
        v16bf bv;
#pragma unroll
        for (int e = 0; e < 16; ++e) {
          const int kc = kc0 + kbB + e + ((e < 8) ? 0 : 8);
          __bf16 v = (__bf16)0.f;
          if (kc < kLim) {
            const int cl = kc / 9;
            const int rr = kc - cl * 9;
            const int dy = rr / 3 - 1;
            const int dx = rr - (rr / 3) * 3 - 1;
            const int yy = y0 + dy, xx = x0 + dx;
            if (yy >= 0 && yy < H && xx >= 0 && xx < W)
              v = smem[cl * RW + (yy - yLo) * W + xx];
          }
          bv[e] = v;
        }
        acc = __builtin_amdgcn_wmma_f32_16x16x32_bf16(
            false, a, false, bv, (short)0, acc, false, false);
      }
    }
    __syncthreads();
  }

  if (!mvalid) return;
  const int y = y0, x = x0;
  const float nw = noise_w[layer];
  uint32_t seed = ((uint32_t)(layer * 31u + (uint32_t)b) * 0x9E3779B9u) ^
                  (((uint32_t)y) << 16) ^ (uint32_t)x;
  const float nz = hash_normal(seed);   // per (b,y,x): broadcast over channels
#pragma unroll
  for (int r = 0; r < 8; ++r) {
    const int co = nt * 16 + r + (lo ? 0 : 8);
    if (co < Co) {
      float v = acc[r] * dmod[(size_t)b * Co + co] + bias[co] + nw * nz;
      v = (v > 0.f) ? v : 0.2f * v;
      xout[(((size_t)b * Co + co) * H + y) * W + x] = (__bf16)v;
    }
  }
}

// ----------------------- ToRGB (1x1 conv, WMMA) ----------------------------

__device__ __forceinline__ v8f rgb_tile(const __bf16* __restrict__ xin,
                                        const __bf16* __restrict__ wgt,
                                        const float* __restrict__ s,
                                        int b, int mt, int Cin, int HW,
                                        int lane)
{
  const int K = Cin;
  const int ksteps = (K + 31) >> 5;
  const int lane15 = lane & 15;
  const bool lo = (lane < 16);
  const int coA = lane15;              // only 3 valid rows
  const bool coAok = (coA < 3);
  const int khalfA = lo ? 0 : 8;
  const int pp = mt * 16 + lane15;
  const int kbB = lo ? 0 : 8;
  const __bf16* wrow = wgt + (size_t)(coAok ? coA : 0) * K;
  const float* srow = s + (size_t)b * Cin;
  const __bf16* xbase = xin + (size_t)b * Cin * HW;
  v8f acc = {0.f, 0.f, 0.f, 0.f, 0.f, 0.f, 0.f, 0.f};
  for (int ks = 0; ks < ksteps; ++ks) {
    const int k0 = ks << 5;
    v16bf a;
    if (coAok && (k0 + 32 <= K)) {
      v8bf a0 = *(const v8bf*)(wrow + k0 + khalfA);
      v8bf a1 = *(const v8bf*)(wrow + k0 + 16 + khalfA);
#pragma unroll
      for (int e = 0; e < 8; ++e) { a[e] = a0[e]; a[8 + e] = a1[e]; }
    } else {
#pragma unroll
      for (int e = 0; e < 16; ++e) {
        const int klocal = (e < 8) ? (khalfA + e) : (8 + khalfA + e);
        const int k = k0 + klocal;
        float v = 0.f;
        if (coAok && k < K) v = (float)wrow[k];
        a[e] = (__bf16)v;
      }
    }
    v16bf bv;
#pragma unroll
    for (int e = 0; e < 16; ++e) {
      const int k = k0 + kbB + e + ((e < 8) ? 0 : 8);
      float v = 0.f;
      if (k < K) v = (float)xbase[(size_t)k * HW + pp] * srow[k];
      bv[e] = (__bf16)v;
    }
    acc = __builtin_amdgcn_wmma_f32_16x16x32_bf16(
        false, a, false, bv, (short)0, acc, false, false);
  }
  return acc;
}

__global__ void __launch_bounds__(256) torgb_wmma(
    const __bf16* __restrict__ xin, const __bf16* __restrict__ wgt,
    const float* __restrict__ s, const float* __restrict__ rgbb,
    const float* __restrict__ rgbbias, float* __restrict__ out,
    int accumulate, int Cin, int H, int W)
{
  const int gtid = blockIdx.x * blockDim.x + threadIdx.x;
  const int wid  = gtid >> 5;
  const int lane = gtid & 31;
  const int HW = H * W;
  const int mtiles = HW >> 4;
  const int total = BATCH * mtiles;
  if (wid >= total) return;
  const int b  = wid / mtiles;
  const int mt = wid - b * mtiles;

  v8f acc = rgb_tile(xin, wgt, s, b, mt, Cin, HW, lane);

  const int p = mt * 16 + (lane & 15);
  const bool lo = (lane < 16);
#pragma unroll
  for (int r = 0; r < 8; ++r) {
    const int co = r + (lo ? 0 : 8);
    if (co < 3) {
      float v = acc[r] + rgbb[co] + rgbbias[co];
      size_t idx = ((size_t)b * 3 + co) * HW + p;
      if (accumulate) out[idx] += v; else out[idx] = v;
    }
  }
}

// ------------------------------- host side ---------------------------------

static inline unsigned gridFor(size_t n) {
  return (unsigned)((n + 255) / 256);
}

extern "C" void kernel_launch(void* const* d_in, const int* in_sizes, int n_in,
                              void* d_out, int out_size, void* d_ws, size_t ws_size,
                              hipStream_t stream)
{
  (void)in_sizes; (void)out_size; (void)ws_size;

  int chin[NLAYERS], chout[NLAYERS];
  { int c = 512;
    for (int i = 0; i < NLAYERS; ++i) {
      int o = (i % 2 == 0 && i > 0) ? c / 2 : c;
      chin[i] = c; chout[i] = o; c = o;
    } }

  const float* z      = (const float*)d_in[0];
  const float* map_w  = (const float*)d_in[1];
  const float* map_b  = (const float*)d_in[2];
  const float* cst    = (const float*)d_in[3];
  const float *conv_w[NLAYERS], *conv_b[NLAYERS], *mod_w[NLAYERS], *mod_b[NLAYERS];
  const float *rgb_w[7], *rgb_b[7], *rgb_mod_w[7], *rgb_mod_b[7], *rgb_bias[7];
  const float* noise_w;

  if (n_in >= 92) {
    for (int i = 0; i < NLAYERS; ++i) {
      conv_w[i] = (const float*)d_in[4 + i];
      conv_b[i] = (const float*)d_in[17 + i];
      mod_w[i]  = (const float*)d_in[30 + i];
      mod_b[i]  = (const float*)d_in[43 + i];
    }
    noise_w = (const float*)d_in[56];
    for (int r = 0; r < 7; ++r) {
      rgb_w[r]     = (const float*)d_in[57 + r];
      rgb_b[r]     = (const float*)d_in[64 + r];
      rgb_mod_w[r] = (const float*)d_in[71 + r];
      rgb_mod_b[r] = (const float*)d_in[78 + r];
      rgb_bias[r]  = (const float*)d_in[85 + r];
    }
  } else {
    const float* cw = (const float*)d_in[4];
    const float* cb = (const float*)d_in[5];
    const float* mw = (const float*)d_in[6];
    const float* mb2 = (const float*)d_in[7];
    noise_w = (const float*)d_in[8];
    const float* rw = (const float*)d_in[9];
    const float* rb = (const float*)d_in[10];
    const float* rmw = (const float*)d_in[11];
    const float* rmb = (const float*)d_in[12];
    const float* rbias = (const float*)d_in[13];
    size_t ocw = 0, ocb = 0, omw = 0, omb = 0;
    for (int i = 0; i < NLAYERS; ++i) {
      conv_w[i] = cw + ocw; ocw += (size_t)chout[i] * chin[i] * 9;
      conv_b[i] = cb + ocb; ocb += chout[i];
      mod_w[i]  = mw + omw; omw += (size_t)chin[i] * WDIM;
      mod_b[i]  = mb2 + omb; omb += chin[i];
    }
    size_t orw = 0, orb = 0, ormw = 0, ormb = 0, orbias = 0;
    for (int r = 0; r < 7; ++r) {
      int ci = chout[2 * r];
      rgb_w[r] = rw + orw;          orw += (size_t)3 * ci;
      rgb_b[r] = rb + orb;          orb += 3;
      rgb_mod_w[r] = rmw + ormw;    ormw += (size_t)ci * WDIM;
      rgb_mod_b[r] = rmb + ormb;    ormb += ci;
      rgb_bias[r] = rbias + orbias; orbias += 3;
    }
  }

  // ---- workspace carve (~80 MB) ----
  char* wp = (char*)d_ws;
  auto carve = [&wp](size_t bytes) -> void* {
    void* r = (void*)wp;
    wp += (bytes + 255) & ~(size_t)255;
    return r;
  };
  float*  wbuf   = (float*)carve((size_t)BATCH * WDIM * 4);
  float*  sbuf   = (float*)carve((size_t)BATCH * WDIM * 4);
  float*  dbuf   = (float*)carve((size_t)BATCH * WDIM * 4);
  float*  srgb   = (float*)carve((size_t)BATCH * WDIM * 4);
  float*  WSb    = (float*)carve((size_t)512 * 512 * 4);
  __bf16* wgtb   = (__bf16*)carve((size_t)512 * 512 * 9 * 2);
  __bf16* wgtrgb = (__bf16*)carve((size_t)3 * 512 * 2);
  const size_t actElems = (size_t)BATCH * 16 * 256 * 256;
  __bf16* xA = (__bf16*)carve(actElems * 2);
  __bf16* xB = (__bf16*)carve(actElems * 2);
  float* rgbA = (float*)carve((size_t)BATCH * 3 * 128 * 128 * 4);
  float* rgbB = (float*)carve((size_t)BATCH * 3 * 128 * 128 * 4);

  mapping_kernel<<<1, 256, 0, stream>>>(z, map_w, map_b, wbuf);
  bcast_const<<<gridFor((size_t)BATCH * 512 * 16), 256, 0, stream>>>(cst, xA, 512 * 16);

  __bf16* cur = xA;
  __bf16* alt = xB;
  float* rgbCur = nullptr;
  int H = 4, W = 4;
  int r = 0;

  for (int i = 0; i < NLAYERS; ++i) {
    const int ci = chin[i];
    const int co = chout[i];
    const bool up = (i % 2 == 0 && i > 0);

    if (up) {
      upsample2x<__bf16><<<gridFor((size_t)BATCH * ci * 4 * H * W), 256, 0, stream>>>(
          cur, alt, ci, H, W);
      __bf16* t2 = cur; cur = alt; alt = t2;
      H *= 2; W *= 2;
    }

    style_kernel<<<gridFor((size_t)BATCH * ci), 256, 0, stream>>>(
        wbuf, mod_w[i], mod_b[i], sbuf, ci);
    prep_weights<<<gridFor((size_t)co * ci), 256, 0, stream>>>(
        conv_w[i], wgtb, WSb, co, ci, 9);
    demod_kernel<<<gridFor((size_t)BATCH * co), 256, 0, stream>>>(
        WSb, sbuf, dbuf, co, ci);
    // fold style into the activations (buffer is consumed by this conv only)
    scale_input<<<gridFor((size_t)BATCH * ci * H * W), 256, 0, stream>>>(
        cur, sbuf, ci, H * W);

    const int mtiles = (H * W) >> 4;
    const int ntiles = (co + 15) >> 4;
    const int mblocks = (mtiles + 7) >> 3;
    modconv_wmma<<<BATCH * ntiles * mblocks, 256, 0, stream>>>(
        cur, wgtb, dbuf, conv_b[i], noise_w, i, alt, ci, co, H, W, mblocks);
    { __bf16* t2 = cur; cur = alt; alt = t2; }

    if (i % 2 == 0) {
      style_kernel<<<gridFor((size_t)BATCH * co), 256, 0, stream>>>(
          wbuf, rgb_mod_w[r], rgb_mod_b[r], srgb, co);
      prep_weights<<<gridFor((size_t)3 * co), 256, 0, stream>>>(
          rgb_w[r], wgtrgb, nullptr, 3, co, 1);

      float* target;
      int accum;
      if (r == 0) {
        target = rgbA;
        accum = 0;
      } else {
        target = (i == NLAYERS - 1) ? (float*)d_out
                                    : ((rgbCur == rgbA) ? rgbB : rgbA);
        upsample2x<float><<<gridFor((size_t)BATCH * 3 * H * W), 256, 0, stream>>>(
            rgbCur, target, 3, H / 2, W / 2);
        accum = 1;
      }
      const size_t rwaves = (size_t)BATCH * ((H * W) >> 4);
      torgb_wmma<<<gridFor(rwaves * 32), 256, 0, stream>>>(
          cur, wgtrgb, srgb, rgb_b[r], rgb_bias[r], target, accum, co, H, W);
      rgbCur = target;
      ++r;
    }
  }
}
